// GNNMutatorPolicy_11647951306787
// MI455X (gfx1250) — matrix operand order repaired and checked by reading, not compiled, not run on hardware
//
#include <hip/hip_runtime.h>
#include <hip/hip_bf16.h>

// ---------------------------------------------------------------------------
// GCN forward on gfx1250 (MI455X): bf16x3-split WMMA GEMMs + atomic scatter.
// N=25000 nodes, E=200000 edges, DIN=HID=512.
// GEMM: fragment-ready LDS layout (ds_load_b128 fragments), pre-split bf16
// hi/lo operands, double-buffered staging via GLOBAL_LOAD_ASYNC_TO_LDS_B128
// (ASYNCcnt-tracked, zero VGPR staging) with a synchronous fallback.
// ---------------------------------------------------------------------------

#define NNODES 25000
#define NEDGES 200000
#define FEAT   512

typedef __attribute__((ext_vector_type(16))) __bf16 v16bf;
typedef __attribute__((ext_vector_type(8)))  float  v8f;
typedef __attribute__((ext_vector_type(4)))  int    v4i;

#if defined(__has_builtin)
#if __has_builtin(__builtin_amdgcn_global_load_async_to_lds_b128)
#define USE_ASYNC_LDS 1
#endif
#endif

#ifdef USE_ASYNC_LDS
// builtin signature (from compiler diagnostic): param0 = global-AS pointer to
// int4, param1 = LDS-AS pointer to int4, then imm offset, imm cpol.
typedef __attribute__((address_space(1))) v4i* glb_v4i_ptr;
typedef __attribute__((address_space(3))) v4i* lds_v4i_ptr;

__device__ __forceinline__ void async_copy16(const __bf16* g, __bf16* l) {
    __builtin_amdgcn_global_load_async_to_lds_b128(
        (glb_v4i_ptr)(void*)g,
        (lds_v4i_ptr)(void*)l,
        0, 0);
}
__device__ __forceinline__ void wait_async0() {
#if __has_builtin(__builtin_amdgcn_s_wait_asynccnt)
    __builtin_amdgcn_s_wait_asynccnt(0);
#else
    asm volatile("s_wait_asynccnt 0x0" ::: "memory");
#endif
}
#endif

// ---------------- small helper kernels ----------------

__global__ void fill_kernel(float* __restrict__ p, float v, long n) {
    long i = (long)blockIdx.x * blockDim.x + threadIdx.x;
    long stride = (long)gridDim.x * blockDim.x;
    for (; i < n; i += stride) p[i] = v;
}

__global__ void deg_accum_kernel(float* __restrict__ deg, const int* __restrict__ dst, int e) {
    int i = blockIdx.x * blockDim.x + threadIdx.x;
    if (i < e) atomicAdd(&deg[dst[i]], 1.0f);
}

__global__ void rsqrt_kernel(float* __restrict__ d, int n) {
    int i = blockIdx.x * blockDim.x + threadIdx.x;
    if (i < n) d[i] = rsqrtf(d[i]);
}

__global__ void edge_norm_kernel(float* __restrict__ nrm, const float* __restrict__ dis,
                                 const int* __restrict__ src, const int* __restrict__ dst, int e) {
    int i = blockIdx.x * blockDim.x + threadIdx.x;
    if (i < e) nrm[i] = dis[src[i]] * dis[dst[i]];
}

// ---------------- fp32 -> bf16 hi/lo split passes --------------------------

__global__ void split_rows_kernel(const float* __restrict__ src, __bf16* __restrict__ hi,
                                  __bf16* __restrict__ lo, long n) {
    long i = (long)blockIdx.x * blockDim.x + threadIdx.x;
    long stride = (long)gridDim.x * blockDim.x;
    for (; i < n; i += stride) {
        float v = src[i];
        __bf16 h = (__bf16)v;
        hi[i] = h;
        lo[i] = (__bf16)(v - (float)h);
    }
}

// weights: [K=512, N=512] row-major -> TRANSPOSED hi/lo bf16 [N, K]
__global__ void split_w_kernel(const float* __restrict__ W, __bf16* __restrict__ hiT,
                               __bf16* __restrict__ loT) {
    int idx = blockIdx.x * blockDim.x + threadIdx.x;   // 512*512 threads
    int k = idx >> 9, c = idx & 511;
    float v = W[idx];
    __bf16 h = (__bf16)v;
    hiT[c * FEAT + k] = h;
    loT[c * FEAT + k] = (__bf16)(v - (float)h);
}

// ---------------- bf16x3 WMMA GEMM:  out[M,512] = A[M,512] @ W[512,512] ----

#define BM 64
#define BN 128
#define BK 32

__global__ __launch_bounds__(256)
void gemm_bf16x3_kernel(const __bf16* __restrict__ Ahi, const __bf16* __restrict__ Alo,
                        const __bf16* __restrict__ BhiT, const __bf16* __restrict__ BloT,
                        float* __restrict__ out, int M) {
    // Fragment-ready LDS: [buf][hi/lo][tile][lane][slot0..15], 32B per lane.
    __shared__ __align__(16) __bf16 sA[2][2][4 * 32 * 16];   // 16 KB
    __shared__ __align__(16) __bf16 sB[2][2][8 * 32 * 16];   // 32 KB

    const int tid   = threadIdx.x;
    const int lane  = tid & 31;
    const int wave  = tid >> 5;      // 0..7
    const int wm    = wave & 3;      // M sub-tile (16 rows each)
    const int wn    = wave >> 2;     // 0..1 (64 cols each)
    const int lrow  = lane & 15;
    const int lhalf = lane >> 4;
    const int m0    = blockIdx.x * BM;
    const int c0    = blockIdx.y * BN;

    // ---- staging chunk math (16B = 8 consecutive K per chunk) ----
    // A: 256 chunks (64 rows x 4 k-chunks), one per thread
    const int am   = tid >> 2;             // row within tile (0..63)
    const int akc  = (tid & 3) * 8;        // k-chunk base (0,8,16,24)
    const int arow = m0 + am;
    // clamp OOB rows to M-1: duplicated data only feeds output rows >= M,
    // which are never stored (needed because async copies can't zero-fill).
    const int arowc = arow < M ? arow : (M - 1);
    const long agsrc = (long)arowc * FEAT + akc;
    const int  aldst = (((am >> 4) * 32 + (am & 15) + 16 * ((akc >> 3) & 1)) * 16
                        + (((akc >> 4) & 1) << 3));
    // B: 512 chunks (128 cols x 4 k-chunks), two per thread
    int  bldst[2];
    long bgsrc[2];
#pragma unroll
    for (int h = 0; h < 2; ++h) {
        int cc = tid + h * 256;
        int c = cc >> 2, kc = (cc & 3) * 8;
        bldst[h] = (((c >> 4) * 32 + (c & 15) + 16 * ((kc >> 3) & 1)) * 16 + (((kc >> 4) & 1) << 3));
        bgsrc[h] = (long)(c0 + c) * FEAT + kc;
    }

#ifdef USE_ASYNC_LDS
    auto stage_async = [&](int buf, int k0) {
        async_copy16(Ahi + agsrc + k0, &sA[buf][0][aldst]);
        async_copy16(Alo + agsrc + k0, &sA[buf][1][aldst]);
#pragma unroll
        for (int h = 0; h < 2; ++h) {
            async_copy16(BhiT + bgsrc[h] + k0, &sB[buf][0][bldst[h]]);
            async_copy16(BloT + bgsrc[h] + k0, &sB[buf][1][bldst[h]]);
        }
    };
#else
    auto stage_sync = [&](int buf, int k0) {
        uint4 a0 = *(const uint4*)(Ahi + agsrc + k0);
        uint4 a1 = *(const uint4*)(Alo + agsrc + k0);
        *(uint4*)(&sA[buf][0][aldst]) = a0;
        *(uint4*)(&sA[buf][1][aldst]) = a1;
#pragma unroll
        for (int h = 0; h < 2; ++h) {
            uint4 b0 = *(const uint4*)(BhiT + bgsrc[h] + k0);
            uint4 b1 = *(const uint4*)(BloT + bgsrc[h] + k0);
            *(uint4*)(&sB[buf][0][bldst[h]]) = b0;
            *(uint4*)(&sB[buf][1][bldst[h]]) = b1;
        }
    };
#endif

    v8f acc[4] = {};                 // 16x64 f32 strip per wave
    const int NT = FEAT / BK;        // 16 k-steps

    auto compute = [&](int buf) {
        // fragments: 32 contiguous bytes per lane -> 2x ds_load_b128 each
        v16bf ahi = *(const v16bf*)&sA[buf][0][(wm * 32 + lane) * 16];
        v16bf alo = *(const v16bf*)&sA[buf][1][(wm * 32 + lane) * 16];
#pragma unroll
        for (int s = 0; s < 4; ++s) {
            const int bbase = ((wn * 4 + s) * 32 + lane) * 16;
            v16bf bhi = *(const v16bf*)&sB[buf][0][bbase];
            v16bf blo = *(const v16bf*)&sB[buf][1][bbase];
            // bf16x3: C += Ahi*Bhi + Ahi*Blo + Alo*Bhi  (f32 accumulate)
            acc[s] = __builtin_amdgcn_wmma_f32_16x16x32_bf16(
                false, ahi, false, bhi, (short)0, acc[s], false, false);
            acc[s] = __builtin_amdgcn_wmma_f32_16x16x32_bf16(
                false, ahi, false, blo, (short)0, acc[s], false, false);
            acc[s] = __builtin_amdgcn_wmma_f32_16x16x32_bf16(
                false, alo, false, bhi, (short)0, acc[s], false, false);
        }
    };

#ifdef USE_ASYNC_LDS
    stage_async(0, 0);
    for (int i = 0; i < NT; ++i) {
        const int buf = i & 1;
        wait_async0();                 // tile i fully in LDS (this wave's copies)
        __syncthreads();               // all waves' copies visible; prev compute done
        if (i + 1 < NT) stage_async(buf ^ 1, (i + 1) * BK);  // overlaps WMMAs below
        compute(buf);
    }
#else
    stage_sync(0, 0);
    for (int i = 0; i < NT; ++i) {
        const int buf = i & 1;
        __syncthreads();
        compute(buf);
        if (i + 1 < NT) stage_sync(buf ^ 1, (i + 1) * BK);
    }
#endif

    // ---- store C (VGPR r: lanes0-15 -> M=r, lanes16-31 -> M=r+8; N = lane&15)
#pragma unroll
    for (int s = 0; s < 4; ++s) {
#pragma unroll
        for (int r = 0; r < 8; ++r) {
            int row = m0 + wm * 16 + lhalf * 8 + r;
            if (row < M)
                out[(long)row * FEAT + c0 + wn * 64 + s * 16 + lrow] = acc[s][r];
        }
    }
}

// ---------------- edge scatter: agg[dst] += hw[src] * norm  (L2-resident) ----

__global__ __launch_bounds__(256)
void scatter_edges_kernel(const float* __restrict__ hw, const int* __restrict__ src,
                          const int* __restrict__ dst, const float* __restrict__ nrm,
                          float* __restrict__ agg) {
    int e = blockIdx.x;
    int s = src[e], d = dst[e];
    float w = nrm[e];
    int f = threadIdx.x;
    float v0 = hw[(long)s * FEAT + f] * w;
    float v1 = hw[(long)s * FEAT + f + 256] * w;
    atomicAdd(&agg[(long)d * FEAT + f], v0);
    atomicAdd(&agg[(long)d * FEAT + f + 256], v1);
}

// ---------------- fused self-loop + bias + ReLU (in-place on agg) ----------

__global__ void epilogue_kernel(float* __restrict__ agg, const float* __restrict__ hw,
                                const float* __restrict__ dis, const float* __restrict__ bias,
                                int n) {
    long idx = (long)blockIdx.x * blockDim.x + threadIdx.x;
    long total = (long)n * FEAT;
    if (idx >= total) return;
    int i = (int)(idx >> 9);
    int f = (int)(idx & (FEAT - 1));
    float d = dis[i];
    float v = agg[idx] + hw[idx] * (d * d) + bias[f];
    agg[idx] = v > 0.0f ? v : 0.0f;
}

// ---------------- heads: two 512->1 dots per node, wave per node ------------

__global__ __launch_bounds__(256)
void heads_kernel(const float* __restrict__ h, const float* __restrict__ Wo,
                  const float* __restrict__ bo, const float* __restrict__ Ww,
                  const float* __restrict__ bw, float* __restrict__ out, int n) {
    int wave = threadIdx.x >> 5;
    int lane = threadIdx.x & 31;
    int i = blockIdx.x * 8 + wave;
    if (i >= n) return;
    float so = 0.0f, sw = 0.0f;
#pragma unroll
    for (int j = 0; j < 16; ++j) {
        int f = lane + j * 32;
        float hv = h[(long)i * FEAT + f];
        so += hv * Wo[f];
        sw += hv * Ww[f];
    }
#pragma unroll
    for (int off = 16; off > 0; off >>= 1) {
        so += __shfl_down(so, off, 32);
        sw += __shfl_down(sw, off, 32);
    }
    if (lane == 0) {
        out[i]     = so + bo[0];
        out[n + i] = sw + bw[0];
    }
}

// ---------------------------------------------------------------------------

static inline size_t align_up(size_t x, size_t a) { return (x + a - 1) & ~(a - 1); }

extern "C" void kernel_launch(void* const* d_in, const int* in_sizes, int n_in,
                              void* d_out, int out_size, void* d_ws, size_t ws_size,
                              hipStream_t stream) {
    (void)in_sizes; (void)n_in; (void)out_size; (void)ws_size;
    const float* x  = (const float*)d_in[0];
    const int*   ei = (const int*)  d_in[1];   // [2, E]: row0 = src, row1 = dst
    const float* W1 = (const float*)d_in[2];
    const float* b1 = (const float*)d_in[3];
    const float* W2 = (const float*)d_in[4];
    const float* b2 = (const float*)d_in[5];
    const float* Wo = (const float*)d_in[6];
    const float* bo = (const float*)d_in[7];
    const float* Ww = (const float*)d_in[8];
    const float* bw = (const float*)d_in[9];
    float* out = (float*)d_out;

    const int* src = ei;
    const int* dst = ei + NEDGES;

    const long NF = (long)NNODES * FEAT;

    // workspace carve (~155.7 MB)
    char* ws = (char*)d_ws;
    size_t off = 0;
    float*  dis  = (float*) (ws + off); off += align_up((size_t)NNODES * 4, 256);
    float*  nrm  = (float*) (ws + off); off += align_up((size_t)NEDGES * 4, 256);
    float*  bufA = (float*) (ws + off); off += align_up((size_t)NF * 4, 256);      // hw
    float*  bufB = (float*) (ws + off); off += align_up((size_t)NF * 4, 256);      // agg / h
    __bf16* Ahi  = (__bf16*)(ws + off); off += align_up((size_t)NF * 2, 256);
    __bf16* Alo  = (__bf16*)(ws + off); off += align_up((size_t)NF * 2, 256);
    __bf16* WhiT = (__bf16*)(ws + off); off += align_up((size_t)FEAT * FEAT * 2, 256);
    __bf16* WloT = (__bf16*)(ws + off);

    dim3 gemm_grid((NNODES + BM - 1) / BM, FEAT / BN);
    const int wsplit_blocks = (FEAT * FEAT) / 256;

    // degree / normalization
    fill_kernel<<<256, 256, 0, stream>>>(dis, 1.0f, NNODES);            // self-loop
    deg_accum_kernel<<<(NEDGES + 255) / 256, 256, 0, stream>>>(dis, dst, NEDGES);
    rsqrt_kernel<<<(NNODES + 255) / 256, 256, 0, stream>>>(dis, NNODES);
    edge_norm_kernel<<<(NEDGES + 255) / 256, 256, 0, stream>>>(nrm, dis, src, dst, NEDGES);

    // ---- layer 1
    split_w_kernel<<<wsplit_blocks, 256, 0, stream>>>(W1, WhiT, WloT);
    split_rows_kernel<<<2048, 256, 0, stream>>>(x, Ahi, Alo, NF);
    gemm_bf16x3_kernel<<<gemm_grid, 256, 0, stream>>>(Ahi, Alo, WhiT, WloT, bufA, NNODES);
    fill_kernel<<<2048, 256, 0, stream>>>(bufB, 0.0f, NF);
    scatter_edges_kernel<<<NEDGES, 256, 0, stream>>>(bufA, src, dst, nrm, bufB);
    epilogue_kernel<<<(int)((NF + 255) / 256), 256, 0, stream>>>(bufB, bufA, dis, b1, NNODES);

    // ---- layer 2 (bufB = h1; split consumed before bufB reused as agg)
    split_w_kernel<<<wsplit_blocks, 256, 0, stream>>>(W2, WhiT, WloT);
    split_rows_kernel<<<2048, 256, 0, stream>>>(bufB, Ahi, Alo, NF);
    gemm_bf16x3_kernel<<<gemm_grid, 256, 0, stream>>>(Ahi, Alo, WhiT, WloT, bufA, NNODES);
    fill_kernel<<<2048, 256, 0, stream>>>(bufB, 0.0f, NF);
    scatter_edges_kernel<<<NEDGES, 256, 0, stream>>>(bufA, src, dst, nrm, bufB);
    epilogue_kernel<<<(int)((NF + 255) / 256), 256, 0, stream>>>(bufB, bufA, dis, b2, NNODES);

    // ---- output heads
    heads_kernel<<<(NNODES + 7) / 8, 256, 0, stream>>>(bufB, Wo, bo, Ww, bw, out, NNODES);
}
